// TransformerWithKMeans_43928925504114
// MI455X (gfx1250) — matrix operand order repaired and checked by reading, not compiled, not run on hardware
//
#include <hip/hip_runtime.h>
#include <hip/hip_bf16.h>

#define DIMM  768
#define NHEAD 12
#define DHEAD 64
#define MLPD  3072
#define SEQ   196
#define BATCH 32
#define ROWS  6272   // BATCH*SEQ
#define KC    64

#ifndef __has_builtin
#define __has_builtin(x) 0
#endif
#if defined(__AMDGCN__) && __has_builtin(__builtin_amdgcn_global_load_async_to_lds_b128) && \
    __has_builtin(__builtin_amdgcn_s_wait_asynccnt)
#define ASYNC_STAGE 1
#else
#define ASYNC_STAGE 0
#endif

typedef __attribute__((ext_vector_type(16))) __bf16 v16bf;
typedef __attribute__((ext_vector_type(8)))  float  v8f;
typedef __attribute__((ext_vector_type(4)))  int    v4i;

#define GLOBAL_AS __attribute__((address_space(1)))
#define LDS_AS    __attribute__((address_space(3)))

struct alignas(16) U4 { unsigned int x, y, z, w; };
union Frag { U4 q[2]; v16bf v; };

// A-matrix 16x32 bf16 fragment (lane=row; K = hi*16 + e, contiguous 16)
__device__ inline v16bf frag_A(const __bf16* p, int hi) {
  Frag f;
  f.q[0] = *(const U4*)(p + hi * 16);
  f.q[1] = *(const U4*)(p + hi * 16 + 8);
  return f.v;
}
// B-matrix 32x16 bf16 fragment (lane=column; K = hi*16 + e, contiguous 16)
__device__ inline v16bf frag_B(const __bf16* p, int hi) {
  Frag f;
  f.q[0] = *(const U4*)(p + hi * 16);
  f.q[1] = *(const U4*)(p + hi * 16 + 8);
  return f.v;
}
__device__ inline v8f wmma_bf16(v16bf a, v16bf b, v8f c) {
  return __builtin_amdgcn_wmma_f32_16x16x32_bf16(false, a, false, b, (short)0, c, false, false);
}
__device__ inline float gelu_exact(float x) {
  return 0.5f * x * (1.0f + erff(x * 0.70710678118654752f));
}

// ---------------------------------------------------------------------------
// bf16 WMMA GEMM with pre-transposed B: out = A[MxK] @ Bt[NxK]^T (+bias)(+res)(gelu?)
// Block tile 64x256, 8 waves (2x4), each wave 32x64 (2x4 WMMA tiles), K-step 32.
// Double-buffered LDS staging; async global->LDS when the toolchain has it.
// Requires M%64==0, N%256==0, K%32==0.
// ---------------------------------------------------------------------------
__global__ void __launch_bounds__(256)
gemm_bf16_k(const __bf16* __restrict__ A, const __bf16* __restrict__ Bt,
            const float* __restrict__ bias, const float* __restrict__ res,
            float* __restrict__ outF, __bf16* __restrict__ outB,
            int M, int N, int Kd, int gelu)
{
  (void)M;
  __shared__ __bf16 sA[2][64][32];     // 8 KB
  __shared__ __bf16 sB[2][256][32];    // 32 KB, B tile stored [n][k] (already transposed)

  const int tid  = threadIdx.x;
  const int lane = tid & 31, wave = tid >> 5;
  const int hi   = lane >> 4, lr = lane & 15;
  const int wm   = wave & 1,  wn = wave >> 1;
  const int blockN = blockIdx.x * 256;
  const int blockM = blockIdx.y * 64;

  const int arow = tid >> 2, ac0 = (tid & 3) << 3;  // A stage: 64 rows x 32 cols, 16B per thread

  v8f acc[2][4] = {};

  auto stage = [&](int kk, int buf) {
    const __bf16* aSrc = A  + (size_t)(blockM + arow) * Kd + kk + ac0;
    const __bf16* bSrc = Bt + (size_t)(blockN + tid)  * Kd + kk;
    __bf16* aDst = &sA[buf][arow][ac0];
    __bf16* bDst = &sB[buf][tid][0];
#if ASYNC_STAGE
    __builtin_amdgcn_global_load_async_to_lds_b128(
        (GLOBAL_AS v4i*)(const void*)aSrc, (LDS_AS v4i*)(void*)aDst, 0, 0);
#pragma unroll
    for (int i = 0; i < 4; ++i)
      __builtin_amdgcn_global_load_async_to_lds_b128(
          (GLOBAL_AS v4i*)(const void*)(bSrc + i * 8), (LDS_AS v4i*)(void*)(bDst + i * 8), 0, 0);
#else
    *(U4*)aDst = *(const U4*)aSrc;
#pragma unroll
    for (int i = 0; i < 4; ++i)
      *(U4*)(bDst + i * 8) = *(const U4*)(bSrc + i * 8);
    if (kk + 32 < Kd) {
      __builtin_prefetch(aSrc + 32, 0, 1);
      __builtin_prefetch(bSrc + 32, 0, 1);
    }
#endif
  };

  int buf = 0;
  stage(0, 0);
  for (int kk = 0; kk < Kd; kk += 32) {
#if ASYNC_STAGE
    __builtin_amdgcn_s_wait_asynccnt(0);
#endif
    __syncthreads();                       // staged tile visible to all waves
    if (kk + 32 < Kd) stage(kk + 32, buf ^ 1);

    v16bf af[2], bfv[4];
#pragma unroll
    for (int mi = 0; mi < 2; ++mi) af[mi]  = frag_A(&sA[buf][wm * 32 + mi * 16 + lr][0], hi);
#pragma unroll
    for (int ni = 0; ni < 4; ++ni) bfv[ni] = frag_B(&sB[buf][wn * 64 + ni * 16 + lr][0], hi);
#pragma unroll
    for (int mi = 0; mi < 2; ++mi)
#pragma unroll
      for (int ni = 0; ni < 4; ++ni)
        acc[mi][ni] = wmma_bf16(af[mi], bfv[ni], acc[mi][ni]);

    __syncthreads();                       // everyone done reading buf before it is re-staged
    buf ^= 1;
  }

#pragma unroll
  for (int mi = 0; mi < 2; ++mi)
#pragma unroll
    for (int ni = 0; ni < 4; ++ni) {
      const int gcol = blockN + wn * 64 + ni * 16 + lr;
      const float bval = bias ? bias[gcol] : 0.0f;
#pragma unroll
      for (int r = 0; r < 8; ++r) {
        const int grow = blockM + wm * 32 + mi * 16 + r + hi * 8;
        float val = acc[mi][ni][r] + bval;
        if (res)  val += res[(size_t)grow * N + gcol];
        if (gelu) val = gelu_exact(val);
        if (outF) outF[(size_t)grow * N + gcol] = val;
        if (outB) outB[(size_t)grow * N + gcol] = (__bf16)val;
      }
    }
}

// ---------------------------------------------------------------------------
// Attention: per (b,h); 4 waves, each wave owns 16-row query tiles.
// S = Q K^T (WMMA) -> softmax in LDS -> O = P V (WMMA). N=196 padded to 208/224.
// ---------------------------------------------------------------------------
__global__ void __launch_bounds__(128)
attention_k(const __bf16* __restrict__ q, const __bf16* __restrict__ k,
            const __bf16* __restrict__ v, __bf16* __restrict__ o)
{
  __shared__ float sS[4][16][224];
  __shared__ float sSum[4][16];

  const int b = blockIdx.x / NHEAD;
  const int h = blockIdx.x % NHEAD;
  const int tid  = threadIdx.x;
  const int lane = tid & 31, w = tid >> 5;
  const int hi   = lane >> 4, lr = lane & 15;
  const size_t base = (size_t)b * SEQ * DIMM + (size_t)h * DHEAD;

  for (int t = 0; t < 4; ++t) {
    const int qt = t * 4 + w;
    const bool valid = (qt < 13);

    if (valid) {
      int qrow = qt * 16 + lr; if (qrow > SEQ - 1) qrow = SEQ - 1;
      const __bf16* qp = q + base + (size_t)qrow * DIMM;
      v16bf qa0 = frag_A(qp, hi);
      v16bf qa1 = frag_A(qp + 32, hi);
      for (int j = 0; j < 13; ++j) {
        const int kr = j * 16 + lr;
        const int krc = (kr > SEQ - 1) ? SEQ - 1 : kr;
        const __bf16* kp = k + base + (size_t)krc * DIMM;
        v16bf kb0 = frag_B(kp, hi);
        v16bf kb1 = frag_B(kp + 32, hi);
        v8f s = {};
        s = wmma_bf16(qa0, kb0, s);
        s = wmma_bf16(qa1, kb1, s);
        const bool masked = (kr >= SEQ);
#pragma unroll
        for (int r = 0; r < 8; ++r)
          sS[w][r + hi * 8][j * 16 + lr] = masked ? -1e30f : s[r] * 0.125f;
      }
    }
    __syncthreads();

    if (valid && lane < 16) {      // softmax: lane = row; in-place f32 -> bf16 P
      float* srow = &sS[w][lr][0];
      __bf16* prow = (__bf16*)srow;
      float m = -1e30f;
      for (int c = 0; c < 208; ++c) m = fmaxf(m, srow[c]);
      float sum = 0.0f;
      for (int c = 0; c < 208; ++c) {
        float p = __expf(srow[c] - m);
        sum += p;
        prow[c] = (__bf16)p;       // 2c <= 4c: safe in-place narrowing
      }
      for (int c = 208; c < 224; ++c) prow[c] = (__bf16)0.0f;
      sSum[w][lr] = sum;
    }
    __syncthreads();

    if (valid) {
      const __bf16* prow = (const __bf16*)&sS[w][lr][0];
      v8f oacc[4] = {};
      for (int ks = 0; ks < 7; ++ks) {
        v16bf pf = frag_A(prow + ks * 32, hi);
#pragma unroll
        for (int dt = 0; dt < 4; ++dt) {
          const int d = dt * 16 + lr;
          v16bf vf;
#pragma unroll
          for (int e = 0; e < 16; ++e) {
            int kr = ks * 32 + hi * 16 + e;
            if (kr > SEQ - 1) kr = SEQ - 1;   // P is 0 there anyway
            vf[e] = v[base + (size_t)kr * DIMM + d];
          }
          oacc[dt] = wmma_bf16(pf, vf, oacc[dt]);
        }
      }
#pragma unroll
      for (int dt = 0; dt < 4; ++dt)
#pragma unroll
        for (int r = 0; r < 8; ++r) {
          const int rr = r + hi * 8;
          const int row = qt * 16 + rr;
          if (row < SEQ)
            o[base + (size_t)row * DIMM + dt * 16 + lr] = (__bf16)(oacc[dt][r] / sSum[w][rr]);
        }
    }
    __syncthreads();
  }
}

// ---------------------------------------------------------------------------
// LayerNorm row kernel (768 = 3 x 256), output bf16
// ---------------------------------------------------------------------------
__global__ void __launch_bounds__(256)
layernorm_k(const float* __restrict__ x, const float* __restrict__ w,
            const float* __restrict__ b, __bf16* __restrict__ out)
{
  __shared__ float red[256];
  const int row = blockIdx.x, tid = threadIdx.x;
  const float* xr = x + (size_t)row * DIMM;
  float v0 = xr[tid], v1 = xr[tid + 256], v2 = xr[tid + 512];
  red[tid] = v0 + v1 + v2; __syncthreads();
  for (int off = 128; off > 0; off >>= 1) { if (tid < off) red[tid] += red[tid + off]; __syncthreads(); }
  const float mean = red[0] * (1.0f / DIMM);
  __syncthreads();
  const float d0 = v0 - mean, d1 = v1 - mean, d2 = v2 - mean;
  red[tid] = d0 * d0 + d1 * d1 + d2 * d2; __syncthreads();
  for (int off = 128; off > 0; off >>= 1) { if (tid < off) red[tid] += red[tid + off]; __syncthreads(); }
  const float rstd = rsqrtf(red[0] * (1.0f / DIMM) + 1e-5f);
  __bf16* orow = out + (size_t)row * DIMM;
  orow[tid]       = (__bf16)(d0 * rstd * w[tid]       + b[tid]);
  orow[tid + 256] = (__bf16)(d1 * rstd * w[tid + 256] + b[tid + 256]);
  orow[tid + 512] = (__bf16)(d2 * rstd * w[tid + 512] + b[tid + 512]);
}

// ---------------------------------------------------------------------------
// k-means: argmin ||p-c||^2 == argmax(p.c - 0.5*||c||^2), scored via WMMA
// ---------------------------------------------------------------------------
__global__ void __launch_bounds__(256)
kmeans_assign_k(const __bf16* __restrict__ pts, const __bf16* __restrict__ cent,
                const float* __restrict__ cnorm, int* __restrict__ labels)
{
  __shared__ float sc[8][16][KC];
  const int tid  = threadIdx.x;
  const int lane = tid & 31, w = tid >> 5;
  const int hi   = lane >> 4, lr = lane & 15;
  const int grow = blockIdx.x * 128 + w * 16 + lr;

  v8f acc[4] = {};
  const __bf16* prow = pts + (size_t)grow * DIMM;
  for (int kk = 0; kk < DIMM; kk += 32) {
    v16bf af = frag_A(prow + kk, hi);
#pragma unroll
    for (int ct = 0; ct < 4; ++ct) {
      v16bf bfv = frag_B(cent + (size_t)(ct * 16 + lr) * DIMM + kk, hi);
      acc[ct] = wmma_bf16(af, bfv, acc[ct]);
    }
  }
#pragma unroll
  for (int ct = 0; ct < 4; ++ct) {
    const int c = ct * 16 + lr;
    const float hn = 0.5f * cnorm[c];
#pragma unroll
    for (int r = 0; r < 8; ++r)
      sc[w][r + hi * 8][c] = acc[ct][r] - hn;
  }
  __syncthreads();
  if (lane < 16) {
    float best = -1e30f; int bi = 0;
    for (int c = 0; c < KC; ++c) {
      const float s = sc[w][lr][c];
      if (s > best) { best = s; bi = c; }
    }
    labels[blockIdx.x * 128 + w * 16 + lr] = bi;
  }
}

__global__ void __launch_bounds__(256)
center_prep_k(const float* __restrict__ centers, __bf16* __restrict__ cent_bf,
              float* __restrict__ cnorm)
{
  __shared__ float red[256];
  const int c = blockIdx.x, tid = threadIdx.x;
  float acc = 0.0f;
#pragma unroll
  for (int i = 0; i < 3; ++i) {
    const int j = tid + i * 256;
    const float x = centers[(size_t)c * DIMM + j];
    cent_bf[(size_t)c * DIMM + j] = (__bf16)x;
    acc += x * x;
  }
  red[tid] = acc; __syncthreads();
  for (int off = 128; off > 0; off >>= 1) { if (tid < off) red[tid] += red[tid + off]; __syncthreads(); }
  if (tid == 0) cnorm[c] = red[0];
}

__global__ void __launch_bounds__(256)
kmeans_accum_k(const float* __restrict__ pts, const int* __restrict__ labels,
               float* __restrict__ sums, float* __restrict__ cnts)
{
  const int i = blockIdx.x;
  const int lab = labels[i];
  for (int j = threadIdx.x; j < DIMM; j += 256)
    atomicAdd(&sums[(size_t)lab * DIMM + j], pts[(size_t)i * DIMM + j]);
  if (threadIdx.x == 0) atomicAdd(&cnts[lab], 1.0f);
}

__global__ void __launch_bounds__(256)
kmeans_update_k(const float* __restrict__ sums, const float* __restrict__ cnts,
                float* __restrict__ centers)
{
  const int c = blockIdx.x;
  const float cnt = cnts[c];
  if (cnt > 0.5f)
    for (int j = threadIdx.x; j < DIMM; j += 256)
      centers[(size_t)c * DIMM + j] = sums[(size_t)c * DIMM + j] / cnt;
}

__global__ void cvt_bf16_k(const float* __restrict__ src, __bf16* __restrict__ dst, long n) {
  long i = (long)blockIdx.x * blockDim.x + threadIdx.x;
  const long stride = (long)gridDim.x * blockDim.x;
  for (; i < n; i += stride) dst[i] = (__bf16)src[i];
}

// transpose-convert: src [Kd][N] f32 -> dst [N][Kd] bf16 (dims % 16 == 0)
__global__ void cvt_t_bf16_k(const float* __restrict__ src, __bf16* __restrict__ dst,
                             int Kd, int N) {
  const int kq = blockIdx.y * 16 + threadIdx.y;
  const int n  = blockIdx.x * 16 + threadIdx.x;
  dst[(size_t)n * Kd + kq] = (__bf16)src[(size_t)kq * N + n];
}

__global__ void zero_f32_k(float* __restrict__ dst, long n) {
  long i = (long)blockIdx.x * blockDim.x + threadIdx.x;
  const long stride = (long)gridDim.x * blockDim.x;
  for (; i < n; i += stride) dst[i] = 0.0f;
}

__global__ void __launch_bounds__(256)
gather_k(const float* __restrict__ projc, const int* __restrict__ labels,
         float* __restrict__ out)
{
  const int i = blockIdx.x;
  const int lab = labels[i];
  for (int j = threadIdx.x; j < DIMM; j += 256)
    out[(size_t)i * DIMM + j] = projc[(size_t)lab * DIMM + j];
}

// ---------------------------------------------------------------------------
extern "C" void kernel_launch(void* const* d_in, const int* in_sizes, int n_in,
                              void* d_out, int out_size, void* d_ws, size_t ws_size,
                              hipStream_t stream)
{
  (void)in_sizes; (void)n_in; (void)out_size; (void)ws_size;

  const float* in_x = (const float*)d_in[0];
  const float* ln1w = (const float*)d_in[1];
  const float* ln1b = (const float*)d_in[2];
  const float* wq   = (const float*)d_in[3];
  const float* wk   = (const float*)d_in[4];
  const float* wvw  = (const float*)d_in[5];
  const float* wo   = (const float*)d_in[6];
  const float* bo   = (const float*)d_in[7];
  const float* ln2w = (const float*)d_in[8];
  const float* ln2b = (const float*)d_in[9];
  const float* w1   = (const float*)d_in[10];
  const float* b1   = (const float*)d_in[11];
  const float* w2   = (const float*)d_in[12];
  const float* b2   = (const float*)d_in[13];
  const float* ctw  = (const float*)d_in[14];
  const float* ctb  = (const float*)d_in[15];
  float* out = (float*)d_out;

  char* ws = (char*)d_ws;
  size_t off = 0;
  auto alloc = [&](size_t bytes) -> void* {
    void* p = ws + off;
    off += (bytes + 255) & ~(size_t)255;
    return p;
  };

  float*  x     = (float*) alloc((size_t)ROWS * DIMM * 4);
  __bf16* h_bf  = (__bf16*)alloc((size_t)ROWS * DIMM * 2);
  __bf16* q_bf  = (__bf16*)alloc((size_t)ROWS * DIMM * 2);
  __bf16* k_bf  = (__bf16*)alloc((size_t)ROWS * DIMM * 2);
  __bf16* v_bf  = (__bf16*)alloc((size_t)ROWS * DIMM * 2);
  __bf16* o_bf  = (__bf16*)alloc((size_t)ROWS * DIMM * 2);
  __bf16* m_bf  = (__bf16*)alloc((size_t)ROWS * MLPD * 2);
  // weights stored TRANSPOSED: [N][K] bf16 per depth slice
  __bf16* wq_bf = (__bf16*)alloc((size_t)4 * DIMM * DIMM * 2);
  __bf16* wk_bf = (__bf16*)alloc((size_t)4 * DIMM * DIMM * 2);
  __bf16* wv_bf = (__bf16*)alloc((size_t)4 * DIMM * DIMM * 2);
  __bf16* wo_bf = (__bf16*)alloc((size_t)4 * DIMM * DIMM * 2);
  __bf16* w1_bf = (__bf16*)alloc((size_t)4 * DIMM * MLPD * 2);
  __bf16* w2_bf = (__bf16*)alloc((size_t)4 * MLPD * DIMM * 2);
  __bf16* ct_bf = (__bf16*)alloc((size_t)DIMM * DIMM * 2);
  float*  centers = (float*) alloc((size_t)KC * DIMM * 4);
  __bf16* cent_bf = (__bf16*)alloc((size_t)KC * DIMM * 2);
  float*  cnorm   = (float*) alloc((size_t)KC * 4);
  float*  sums    = (float*) alloc((size_t)(KC * DIMM + KC) * 4);
  float*  cnts    = sums + (size_t)KC * DIMM;
  float*  projc   = (float*) alloc((size_t)KC * DIMM * 4);
  int*    labels  = (int*)   alloc((size_t)ROWS * 4);

  (void)hipMemcpyAsync(x, in_x, (size_t)ROWS * DIMM * 4, hipMemcpyDeviceToDevice, stream);

  // one-time transpose-convert of all weights to bf16 [N][K]
  const dim3 tb(16, 16);
  for (int d = 0; d < 4; ++d) {
    const size_t w768 = (size_t)d * DIMM * DIMM;
    const size_t wmlp = (size_t)d * DIMM * MLPD;
    cvt_t_bf16_k<<<dim3(DIMM / 16, DIMM / 16), tb, 0, stream>>>(wq  + w768, wq_bf + w768, DIMM, DIMM);
    cvt_t_bf16_k<<<dim3(DIMM / 16, DIMM / 16), tb, 0, stream>>>(wk  + w768, wk_bf + w768, DIMM, DIMM);
    cvt_t_bf16_k<<<dim3(DIMM / 16, DIMM / 16), tb, 0, stream>>>(wvw + w768, wv_bf + w768, DIMM, DIMM);
    cvt_t_bf16_k<<<dim3(DIMM / 16, DIMM / 16), tb, 0, stream>>>(wo  + w768, wo_bf + w768, DIMM, DIMM);
    cvt_t_bf16_k<<<dim3(MLPD / 16, DIMM / 16), tb, 0, stream>>>(w1  + wmlp, w1_bf + wmlp, DIMM, MLPD);
    cvt_t_bf16_k<<<dim3(DIMM / 16, MLPD / 16), tb, 0, stream>>>(w2  + wmlp, w2_bf + wmlp, MLPD, DIMM);
  }
  cvt_t_bf16_k<<<dim3(DIMM / 16, DIMM / 16), tb, 0, stream>>>(ctw, ct_bf, DIMM, DIMM);

  const dim3 g768(DIMM / 256, ROWS / 64);
  const dim3 gmlp(MLPD / 256, ROWS / 64);

  for (int d = 0; d < 4; ++d) {
    const size_t wofs = (size_t)d * DIMM * DIMM;
    layernorm_k<<<ROWS, 256, 0, stream>>>(x, ln1w + d * DIMM, ln1b + d * DIMM, h_bf);
    gemm_bf16_k<<<g768, 256, 0, stream>>>(h_bf, wq_bf + wofs, nullptr, nullptr, nullptr, q_bf, ROWS, DIMM, DIMM, 0);
    gemm_bf16_k<<<g768, 256, 0, stream>>>(h_bf, wk_bf + wofs, nullptr, nullptr, nullptr, k_bf, ROWS, DIMM, DIMM, 0);
    gemm_bf16_k<<<g768, 256, 0, stream>>>(h_bf, wv_bf + wofs, nullptr, nullptr, nullptr, v_bf, ROWS, DIMM, DIMM, 0);
    attention_k<<<BATCH * NHEAD, 128, 0, stream>>>(q_bf, k_bf, v_bf, o_bf);
    gemm_bf16_k<<<g768, 256, 0, stream>>>(o_bf, wo_bf + wofs, bo + d * DIMM, x, x, nullptr, ROWS, DIMM, DIMM, 0);
    layernorm_k<<<ROWS, 256, 0, stream>>>(x, ln2w + d * DIMM, ln2b + d * DIMM, h_bf);
    gemm_bf16_k<<<gmlp, 256, 0, stream>>>(h_bf, w1_bf + (size_t)d * DIMM * MLPD, b1 + d * MLPD,
                                          nullptr, nullptr, m_bf, ROWS, MLPD, DIMM, 1);
    gemm_bf16_k<<<g768, 256, 0, stream>>>(m_bf, w2_bf + (size_t)d * MLPD * DIMM, b2 + d * DIMM,
                                          x, x, nullptr, ROWS, DIMM, MLPD, 0);
  }

  // k-means on final activations
  cvt_bf16_k<<<2048, 256, 0, stream>>>(x, h_bf, (long)ROWS * DIMM);   // points bf16 (reuse h_bf)
  (void)hipMemcpyAsync(centers, x, (size_t)KC * DIMM * 4, hipMemcpyDeviceToDevice, stream);

  for (int it = 0; it < 10; ++it) {
    center_prep_k<<<KC, 256, 0, stream>>>(centers, cent_bf, cnorm);
    kmeans_assign_k<<<ROWS / 128, 256, 0, stream>>>(h_bf, cent_bf, cnorm, labels);
    zero_f32_k<<<256, 256, 0, stream>>>(sums, (long)KC * DIMM + KC);
    kmeans_accum_k<<<ROWS, 256, 0, stream>>>(x, labels, sums, cnts);
    kmeans_update_k<<<KC, 256, 0, stream>>>(sums, cnts, centers);
  }
  center_prep_k<<<KC, 256, 0, stream>>>(centers, cent_bf, cnorm);
  kmeans_assign_k<<<ROWS / 128, 256, 0, stream>>>(h_bf, cent_bf, cnorm, labels);

  // out = centers[labels] @ ct_w + ct_b  ==  (centers @ ct_w + ct_b)[labels]
  const dim3 gct(DIMM / 256, KC / 64);
  gemm_bf16_k<<<gct, 256, 0, stream>>>(cent_bf, ct_bf, ctb, nullptr, projc, nullptr, KC, DIMM, DIMM, 0);
  gather_k<<<ROWS, 256, 0, stream>>>(projc, labels, out);
}
